// GatedCrossAttention_27273042329697
// MI455X (gfx1250) — matrix operand order repaired and checked by reading, not compiled, not run on hardware
//
#include <hip/hip_runtime.h>
#include <hip/hip_bf16.h>

// ---------------- problem constants ----------------
#define DIMV       1024
#define NHEADS     16
#define DHEAD      64
#define INNERV     1024
#define BATCH      2
#define SQLEN      2048
#define SKLEN      4096
#define QK_SCALE   0.125f   // 64^-0.5

typedef __bf16 bf16;
typedef __attribute__((ext_vector_type(16))) __bf16 bf16x16;
typedef __attribute__((ext_vector_type(8)))  __bf16 bf16x8;
typedef __attribute__((ext_vector_type(8)))  float  f32x8;
typedef __attribute__((ext_vector_type(4)))  unsigned int u32x4;
typedef __attribute__((ext_vector_type(8)))  int  i32x8;
typedef __attribute__((ext_vector_type(4)))  int  i32x4;

union BF16Frag { bf16x16 v; bf16x8 h[2]; };

__device__ __forceinline__ f32x8 f32x8_zero() {
  f32x8 z;
#pragma unroll
  for (int i = 0; i < 8; ++i) z[i] = 0.0f;
  return z;
}

__device__ __forceinline__ f32x8 wmma_bf16(bf16x16 a, bf16x16 b, f32x8 c) {
  return __builtin_amdgcn_wmma_f32_16x16x32_bf16(
      false, a, false, b, (short)0, c, false, false);
}

// A fragment: row-major A[M x K]; lane holds row (lane&15), half=lane>>4
//   elems 0..15 = K (k0 + half*16 + e)
__device__ __forceinline__ bf16x16 load_a_frag(const bf16* A, int lda, int row0, int k0) {
  const int lane = threadIdx.x & 31;
  const int r = lane & 15, half = lane >> 4;
  const bf16* base = A + (long)(row0 + r) * lda + k0;
  BF16Frag f;
  f.h[0] = *(const bf16x8*)(base + half * 16);
  f.h[1] = *(const bf16x8*)(base + half * 16 + 8);
  return f.v;
}

// B fragment from B-transposed storage (BT row-major [N x K]):
//   lane holds col (lane&15); elems 0..15 = K (k0 + half*16 + e)
__device__ __forceinline__ bf16x16 load_bT_frag(const bf16* BT, int ldb, int col0, int k0) {
  const int lane = threadIdx.x & 31;
  const int n = lane & 15, half = lane >> 4;
  return *(const bf16x16*)(BT + (long)(col0 + n) * ldb + k0 + half * 16);
}

// ---------------- Tensor Data Mover: 2-D tile -> LDS ----------------
// D# groups per CDNA5 ISA (CDNA5 ISA).
__device__ __forceinline__ void tdm_load_2d(unsigned lds_byte_off,
                                            const void* gptr,
                                            unsigned tensor_d0, unsigned tensor_d1,
                                            unsigned tile_d0,   unsigned tile_d1,
                                            unsigned long stride0_elems) {
  const unsigned long ga = (unsigned long)gptr;
  u32x4 g0;
  g0[0] = 1u;                                              // count=1, user mode
  g0[1] = lds_byte_off;                                    // LDS address (bytes)
  g0[2] = (unsigned)(ga & 0xFFFFFFFFu);                    // global addr [31:0]
  g0[3] = (unsigned)((ga >> 32) & 0x1FFFFFFu) | (2u << 30);// addr[56:32] | type=2
  i32x8 g1;
  g1[0] = (int)(1u << 16);                                 // wg_mask=0, data_size=1(2B)
  g1[1] = (int)((tensor_d0 & 0xFFFFu) << 16);              // atomic_addr=0 | dim0.lo16
  g1[2] = (int)(((tensor_d0 >> 16) & 0xFFFFu) | ((tensor_d1 & 0xFFFFu) << 16));
  g1[3] = (int)(((tensor_d1 >> 16) & 0xFFFFu) | ((tile_d0 & 0xFFFFu) << 16));
  g1[4] = (int)(tile_d1 & 0xFFFFu);                        // tile_dim1 | tile_dim2=0
  g1[5] = (int)(stride0_elems & 0xFFFFFFFFu);              // dim0 stride [31:0]
  g1[6] = (int)((stride0_elems >> 32) & 0xFFFFu);          // stride[47:32] | stride1=0
  g1[7] = 0;
  const i32x4 z4 = {0, 0, 0, 0};
#if __clang_major__ >= 23
  const i32x8 z8 = {0, 0, 0, 0, 0, 0, 0, 0};
  __builtin_amdgcn_tensor_load_to_lds(g0, g1, z4, z4, z8, 0);
#else
  __builtin_amdgcn_tensor_load_to_lds(g0, g1, z4, z4, 0);
#endif
}

// ---------------- prep kernels ----------------
__global__ void transpose_to_bf16_kernel(const float* __restrict__ src,
                                         bf16* __restrict__ dst, int K, int N) {
  long idx = (long)blockIdx.x * blockDim.x + threadIdx.x;
  if (idx >= (long)K * N) return;
  int n = (int)(idx / K);
  int k = (int)(idx % K);
  dst[idx] = (bf16)src[(long)k * N + n];
}

__global__ void to_bf16_kernel(const float* __restrict__ src,
                               bf16* __restrict__ dst, long n) {
  long idx = (long)blockIdx.x * blockDim.x + threadIdx.x;
  if (idx < n) dst[idx] = (bf16)src[idx];
}

__global__ void layernorm_kernel(const float* __restrict__ x,
                                 const float* __restrict__ gamma,
                                 const float* __restrict__ beta,
                                 bf16* __restrict__ out) {
  __shared__ float s1[256];
  __shared__ float s2[256];
  const int row = blockIdx.x;
  const int tid = threadIdx.x;
  const float* xr = x + (long)row * DIMV;
  float a = 0.f, b = 0.f;
#pragma unroll
  for (int i = tid; i < DIMV; i += 256) { float v = xr[i]; a += v; b += v * v; }
  s1[tid] = a; s2[tid] = b;
  __syncthreads();
  for (int st = 128; st > 0; st >>= 1) {
    if (tid < st) { s1[tid] += s1[tid + st]; s2[tid] += s2[tid + st]; }
    __syncthreads();
  }
  const float mean = s1[0] * (1.0f / DIMV);
  const float var  = s2[0] * (1.0f / DIMV) - mean * mean;
  const float rstd = rsqrtf(var + 1e-5f);
  bf16* orow = out + (long)row * DIMV;
#pragma unroll
  for (int i = tid; i < DIMV; i += 256)
    orow[i] = (bf16)((xr[i] - mean) * rstd * gamma[i] + beta[i]);
}

// ---------------- register-blocked WMMA GEMM core: 32x64 C per wave ----------------
__device__ __forceinline__ void gemm_32x64(const bf16* A, int lda, int row0,
                                           const bf16* BT, int ldb, int col0,
                                           int Kdim, f32x8 acc[2][4]) {
#pragma unroll
  for (int i = 0; i < 2; ++i)
#pragma unroll
    for (int j = 0; j < 4; ++j) acc[i][j] = f32x8_zero();
  for (int k0 = 0; k0 < Kdim; k0 += 32) {
    bf16x16 af[2], bfr[4];
#pragma unroll
    for (int i = 0; i < 2; ++i) af[i] = load_a_frag(A, lda, row0 + 16 * i, k0);
#pragma unroll
    for (int j = 0; j < 4; ++j) bfr[j] = load_bT_frag(BT, ldb, col0 + 16 * j, k0);
#pragma unroll
    for (int i = 0; i < 2; ++i)
#pragma unroll
      for (int j = 0; j < 4; ++j) acc[i][j] = wmma_bf16(af[i], bfr[j], acc[i][j]);
  }
}

// q = xn @ w_q (scaled); q[b][h][sq][d] bf16
__global__ void __launch_bounds__(256)
gemm_q_kernel(const bf16* __restrict__ XN, const bf16* __restrict__ WQT,
              bf16* __restrict__ Q) {
  const int gwave = (int)((blockIdx.x * blockDim.x + threadIdx.x) >> 5);
  const int TN = INNERV / 64;                    // 16
  const int tm = gwave / TN, tn = gwave % TN;
  const int row0 = tm * 32, col0 = tn * 64;

  f32x8 acc[2][4];
  gemm_32x64(XN, DIMV, row0, WQT, DIMV, col0, DIMV, acc);

  const int lane = threadIdx.x & 31;
  const int nn = lane & 15, hi = lane >> 4;
#pragma unroll
  for (int j = 0; j < 4; ++j) {
    const int col = col0 + j * 16 + nn;
    const int head = col >> 6, d = col & 63;
#pragma unroll
    for (int i = 0; i < 2; ++i)
#pragma unroll
      for (int r = 0; r < 8; ++r) {
        const int m  = row0 + i * 16 + r + 8 * hi;
        const int bb = m >> 11;
        const int sq = m & (SQLEN - 1);
        Q[(((long)(bb * NHEADS + head) * SQLEN + sq) << 6) + d] =
            (bf16)(acc[i][j][r] * QK_SCALE);
      }
  }
}

// kv = context @ w_kv -> k[b][h][sk][d], vT[b][h][d][sk]
__global__ void __launch_bounds__(256)
gemm_kv_kernel(const bf16* __restrict__ CTX, const bf16* __restrict__ WKVT,
               bf16* __restrict__ Kb, bf16* __restrict__ VT) {
  const int gwave = (int)((blockIdx.x * blockDim.x + threadIdx.x) >> 5);
  const int TN = (2 * INNERV) / 64;              // 32
  const int tm = gwave / TN, tn = gwave % TN;
  const int row0 = tm * 32, col0 = tn * 64;

  f32x8 acc[2][4];
  gemm_32x64(CTX, DIMV, row0, WKVT, DIMV, col0, DIMV, acc);

  const int lane = threadIdx.x & 31;
  const int nn = lane & 15, hi = lane >> 4;
#pragma unroll
  for (int j = 0; j < 4; ++j) {
    const int col = col0 + j * 16 + nn;
    if (col < INNERV) {
      const int head = col >> 6, d = col & 63;
#pragma unroll
      for (int i = 0; i < 2; ++i)
#pragma unroll
        for (int r = 0; r < 8; ++r) {
          const int m  = row0 + i * 16 + r + 8 * hi;
          const int bb = m >> 12;
          const int sk = m & (SKLEN - 1);
          Kb[(((long)(bb * NHEADS + head) * SKLEN + sk) << 6) + d] = (bf16)acc[i][j][r];
        }
    } else {
      const int c = col - INNERV;
      const int head = c >> 6, d = c & 63;
#pragma unroll
      for (int i = 0; i < 2; ++i)
#pragma unroll
        for (int r = 0; r < 8; ++r) {
          const int m  = row0 + i * 16 + r + 8 * hi;
          const int bb = m >> 12;
          const int sk = m & (SKLEN - 1);
          VT[(((long)((bb * NHEADS + head) << 6) + d) * SKLEN) + sk] = (bf16)acc[i][j][r];
        }
    }
  }
}

// out = O @ w_out + b_out, times tanh(gate); f32 output
__global__ void __launch_bounds__(256)
gemm_out_kernel(const bf16* __restrict__ OB, const bf16* __restrict__ WOT,
                const float* __restrict__ bout, const float* __restrict__ gate,
                float* __restrict__ out) {
  const int gwave = (int)((blockIdx.x * blockDim.x + threadIdx.x) >> 5);
  const int TN = DIMV / 64;                      // 16
  const int tm = gwave / TN, tn = gwave % TN;
  const int row0 = tm * 32, col0 = tn * 64;

  f32x8 acc[2][4];
  gemm_32x64(OB, INNERV, row0, WOT, INNERV, col0, INNERV, acc);

  const float gt = tanhf(gate[0]);
  const int lane = threadIdx.x & 31;
  const int nn = lane & 15, hi = lane >> 4;
#pragma unroll
  for (int j = 0; j < 4; ++j) {
    const int col = col0 + j * 16 + nn;
    const float bias = bout[col];
#pragma unroll
    for (int i = 0; i < 2; ++i)
#pragma unroll
      for (int r = 0; r < 8; ++r) {
        const int m = row0 + i * 16 + r + 8 * hi;
        out[(long)m * DIMV + col] = (acc[i][j][r] + bias) * gt;
      }
  }
}

// ---------------- flash attention: 1 wave = (b,h, 32 q rows) ----------------
// K/V tiles staged to LDS by the Tensor Data Mover, double-buffered (ping-pong):
// DMA for step s+32 is issued before computing step s; s_wait_tensorcnt(2)
// guarantees the current pair has landed (same-wave TDM ops retire in order).
#define LDSK_ELEMS (32 * DHEAD)     // [32 sk][64 d]
#define LDSV_ELEMS (DHEAD * 32)     // [64 d][32 sk]
#define LDSBUF_ELEMS (LDSK_ELEMS + LDSV_ELEMS)
#define LDSP_ELEMS (2 * 16 * 32)    // two 16x32 P tiles

__global__ void __launch_bounds__(32)
flash_attn_kernel(const bf16* __restrict__ Q,    // [B,H,SQ,DH] (pre-scaled)
                  const bf16* __restrict__ Kb,   // [B,H,SK,DH]
                  const bf16* __restrict__ VT,   // [B,H,DH,SK]
                  bf16* __restrict__ OB) {       // [B*SQ, INNER]
  __shared__ __align__(16) bf16 smem[2 * LDSBUF_ELEMS + LDSP_ELEMS];
  bf16* ldsP = smem + 2 * LDSBUF_ELEMS;

  const int bid = blockIdx.x;
  const int qt  = bid & (SQLEN / 32 - 1);            // 64 q blocks of 32 rows
  const int bh  = bid >> 6;                          // 0..31 (= b*16 + h)
  const int lane = threadIdx.x & 31;
  const int nn = lane & 15, hi = lane >> 4;

  const bf16* Qbase = Q  + ((long)bh * SQLEN + qt * 32) * DHEAD;
  const bf16* Kbase = Kb + (long)bh * SKLEN * DHEAD;
  const bf16* Vbase = VT + (long)bh * DHEAD * SKLEN;

  bf16x16 qa[2][2];
#pragma unroll
  for (int i = 0; i < 2; ++i) {
    qa[i][0] = load_a_frag(Qbase, DHEAD, 16 * i, 0);
    qa[i][1] = load_a_frag(Qbase, DHEAD, 16 * i, 32);
  }

  f32x8 o[2][4];
  float mrow[2][8], lrow[2][8];
#pragma unroll
  for (int i = 0; i < 2; ++i) {
#pragma unroll
    for (int t = 0; t < 4; ++t) o[i][t] = f32x8_zero();
#pragma unroll
    for (int r = 0; r < 8; ++r) { mrow[i][r] = -1e30f; lrow[i][r] = 0.0f; }
  }

  // ---- preload tile 0 into buffer 0 ----
  tdm_load_2d(0u, Kbase, DHEAD, 32, DHEAD, 32, DHEAD);
  tdm_load_2d((unsigned)(LDSK_ELEMS * 2), Vbase, 32, DHEAD, 32, DHEAD, SKLEN);

  for (int s0 = 0; s0 < SKLEN; s0 += 32) {
    const int cur = (s0 >> 5) & 1;
    const int nxt = cur ^ 1;
    if (s0 + 32 < SKLEN) {
      const unsigned nbase = (unsigned)(nxt * LDSBUF_ELEMS * 2);
      tdm_load_2d(nbase, Kbase + (long)(s0 + 32) * DHEAD, DHEAD, 32, DHEAD, 32, DHEAD);
      tdm_load_2d(nbase + (unsigned)(LDSK_ELEMS * 2), Vbase + (s0 + 32),
                  32, DHEAD, 32, DHEAD, SKLEN);
      __builtin_amdgcn_s_wait_tensorcnt(2);   // current pair has landed
    } else {
      __builtin_amdgcn_s_wait_tensorcnt(0);
    }
    const bf16* ldsK = smem + cur * LDSBUF_ELEMS;
    const bf16* ldsV = smem + cur * LDSBUF_ELEMS + LDSK_ELEMS;

    // ---- K B-fragments from LDS (shared by both q-tiles) ----
    bf16x16 bk[2][2];  // [16-col tile][d chunk]
#pragma unroll
    for (int t = 0; t < 2; ++t)
#pragma unroll
      for (int c = 0; c < 2; ++c)
        bk[t][c] = *(const bf16x16*)(ldsK + (t * 16 + nn) * DHEAD + c * 32 + hi * 16);

    // ---- scores + online softmax per q-tile ----
#pragma unroll
    for (int i = 0; i < 2; ++i) {
      f32x8 sa = f32x8_zero(), sb = f32x8_zero();
      sa = wmma_bf16(qa[i][0], bk[0][0], sa);
      sa = wmma_bf16(qa[i][1], bk[0][1], sa);
      sb = wmma_bf16(qa[i][0], bk[1][0], sb);
      sb = wmma_bf16(qa[i][1], bk[1][1], sb);

      float scl[8];
#pragma unroll
      for (int r = 0; r < 8; ++r) {
        float mx = fmaxf(sa[r], sb[r]);
#pragma unroll
        for (int msk = 1; msk < 16; msk <<= 1) mx = fmaxf(mx, __shfl_xor(mx, msk, 32));
        const float mnew = fmaxf(mrow[i][r], mx);
        scl[r] = __expf(mrow[i][r] - mnew);
        mrow[i][r] = mnew;
        const float pa = __expf(sa[r] - mnew);
        const float pb = __expf(sb[r] - mnew);
        float rs = pa + pb;
#pragma unroll
        for (int msk = 1; msk < 16; msk <<= 1) rs += __shfl_xor(rs, msk, 32);
        lrow[i][r] = lrow[i][r] * scl[r] + rs;
        sa[r] = pa; sb[r] = pb;
      }
#pragma unroll
      for (int r = 0; r < 8; ++r) {
        o[i][0][r] *= scl[r]; o[i][1][r] *= scl[r];
        o[i][2][r] *= scl[r]; o[i][3][r] *= scl[r];
      }
      // write P tile i to LDS (D-layout -> row-major)
      bf16* pb_ = ldsP + i * (16 * 32);
#pragma unroll
      for (int r = 0; r < 8; ++r) {
        const int m = r + 8 * hi;
        pb_[m * 32 + nn]      = (bf16)sa[r];
        pb_[m * 32 + nn + 16] = (bf16)sb[r];
      }
    }
    __syncthreads();

    // ---- P A-fragments and shared V B-fragments ----
    BF16Frag pf[2];
#pragma unroll
    for (int i = 0; i < 2; ++i) {
      const bf16* pb_ = ldsP + i * (16 * 32);
      pf[i].h[0] = *(const bf16x8*)(pb_ + nn * 32 + hi * 16);
      pf[i].h[1] = *(const bf16x8*)(pb_ + nn * 32 + hi * 16 + 8);
    }
    bf16x16 vf[4];
#pragma unroll
    for (int t = 0; t < 4; ++t)
      vf[t] = *(const bf16x16*)(ldsV + (t * 16 + nn) * 32 + hi * 16);

    // ---- O(32x64) += P(32x32) * V(32x64) ----
#pragma unroll
    for (int i = 0; i < 2; ++i)
#pragma unroll
      for (int t = 0; t < 4; ++t)
        o[i][t] = wmma_bf16(pf[i].v, vf[t], o[i][t]);
    __syncthreads();   // P tile reuse next iteration
  }

  // ---- finalize ----
  const int bb = bh >> 4;
  const int h  = bh & 15;
#pragma unroll
  for (int i = 0; i < 2; ++i)
#pragma unroll
    for (int r = 0; r < 8; ++r) {
      const float inv = 1.0f / lrow[i][r];
      const long grow = (long)bb * SQLEN + qt * 32 + i * 16 + r + 8 * hi;
      bf16* op = OB + grow * INNERV + (h << 6) + nn;
      op[0]  = (bf16)(o[i][0][r] * inv);
      op[16] = (bf16)(o[i][1][r] * inv);
      op[32] = (bf16)(o[i][2][r] * inv);
      op[48] = (bf16)(o[i][3][r] * inv);
    }
}

// ---------------- workspace layout (bytes) ----------------
#define OFF_XN    ((size_t)0)
#define OFF_CTX   ((size_t)8  << 20)
#define OFF_WQT   ((size_t)24 << 20)
#define OFF_WKVT  ((size_t)26 << 20)
#define OFF_WOUTT ((size_t)30 << 20)
#define OFF_Q     ((size_t)32 << 20)
#define OFF_K     ((size_t)40 << 20)
#define OFF_VT    ((size_t)56 << 20)
#define OFF_OB    ((size_t)72 << 20)
// total: 80 MiB

extern "C" void kernel_launch(void* const* d_in, const int* in_sizes, int n_in,
                              void* d_out, int out_size, void* d_ws, size_t ws_size,
                              hipStream_t stream) {
  (void)in_sizes; (void)n_in; (void)out_size; (void)ws_size;
  const float* x        = (const float*)d_in[0];
  const float* context  = (const float*)d_in[1];
  const float* ln_gamma = (const float*)d_in[2];
  const float* ln_beta  = (const float*)d_in[3];
  const float* w_q      = (const float*)d_in[4];
  const float* w_kv     = (const float*)d_in[5];
  const float* w_out    = (const float*)d_in[6];
  const float* b_out    = (const float*)d_in[7];
  const float* gate     = (const float*)d_in[8];
  float* out = (float*)d_out;

  char* ws = (char*)d_ws;
  bf16* XN   = (bf16*)(ws + OFF_XN);
  bf16* CTX  = (bf16*)(ws + OFF_CTX);
  bf16* WQT  = (bf16*)(ws + OFF_WQT);
  bf16* WKVT = (bf16*)(ws + OFF_WKVT);
  bf16* WOT  = (bf16*)(ws + OFF_WOUTT);
  bf16* Qb   = (bf16*)(ws + OFF_Q);
  bf16* Kb   = (bf16*)(ws + OFF_K);
  bf16* VTb  = (bf16*)(ws + OFF_VT);
  bf16* OB   = (bf16*)(ws + OFF_OB);

  transpose_to_bf16_kernel<<<(DIMV * INNERV) / 256, 256, 0, stream>>>(w_q, WQT, DIMV, INNERV);
  transpose_to_bf16_kernel<<<(DIMV * 2 * INNERV) / 256, 256, 0, stream>>>(w_kv, WKVT, DIMV, 2 * INNERV);
  transpose_to_bf16_kernel<<<(INNERV * DIMV) / 256, 256, 0, stream>>>(w_out, WOT, INNERV, DIMV);
  to_bf16_kernel<<<(BATCH * SKLEN * DIMV) / 256, 256, 0, stream>>>(context, CTX, (long)BATCH * SKLEN * DIMV);
  layernorm_kernel<<<BATCH * SQLEN, 256, 0, stream>>>(x, ln_gamma, ln_beta, XN);

  gemm_q_kernel<<<((BATCH * SQLEN / 32) * (INNERV / 64)) / 8, 256, 0, stream>>>(XN, WQT, Qb);
  gemm_kv_kernel<<<((BATCH * SKLEN / 32) * (2 * INNERV / 64)) / 8, 256, 0, stream>>>(CTX, WKVT, Kb, VTb);

  flash_attn_kernel<<<BATCH * NHEADS * (SQLEN / 32), 32, 0, stream>>>(Qb, Kb, VTb, OB);

  gemm_out_kernel<<<((BATCH * SQLEN / 32) * (DIMV / 64)) / 8, 256, 0, stream>>>(OB, WOT, b_out, gate, out);
}
